// HadamardMLPPairwise_29669634081455
// MI455X (gfx1250) — matrix operand order repaired and checked by reading, not compiled, not run on hardware
//
#include <hip/hip_runtime.h>
#include <hip/hip_bf16.h>

#define BATCH 4
#define NT    2048
#define MS    2048
#define DIM   128
#define HID   128
#define ASTRIDE 68   // dwords per bf16-pair row of A_i: 64 pairs + 4 pad (16B-aligned rows, spreads banks)

typedef __attribute__((ext_vector_type(16))) __bf16 v16bf;
typedef __attribute__((ext_vector_type(8)))  float  v8f;

union FragBF {
    unsigned int u[8];
    uint4        q[2];
    v16bf        v;
};

// Truncate two f32 to bf16 and pack into one dword with a single v_perm_b32.
// dst = { hi[31:16], lo[31:16] }  (perm sel 0x07060302 over {hi, lo})
__device__ __forceinline__ unsigned int pack_bf16(float lo, float hi) {
    return __builtin_amdgcn_perm(__float_as_uint(hi), __float_as_uint(lo), 0x07060302u);
}

// Load one 16x128 bf16 A-operand set (4 K-steps of 32) from a row of source_val.
// ISA A-frag layout: VGPR v<4 -> K = 2v + 8*hi ; v>=4 -> K = 16 + 2(v-4) + 8*hi.
__device__ __forceinline__ void load_afrags(const float* __restrict__ srow, int hi, FragBF af[4]) {
    #pragma unroll
    for (int kk = 0; kk < 4; ++kk) {
        const float* p = srow + 32 * kk + 8 * hi;      // 32B aligned
        float4 x0 = *(const float4*)(p);               // K = 8hi + 0..3
        float4 x1 = *(const float4*)(p + 4);           // K = 8hi + 4..7
        float4 y0 = *(const float4*)(p + 16);          // K = 16 + 8hi + 0..3
        float4 y1 = *(const float4*)(p + 20);          // K = 16 + 8hi + 4..7
        af[kk].u[0] = pack_bf16(x0.x, x0.y);
        af[kk].u[1] = pack_bf16(x0.z, x0.w);
        af[kk].u[2] = pack_bf16(x1.x, x1.y);
        af[kk].u[3] = pack_bf16(x1.z, x1.w);
        af[kk].u[4] = pack_bf16(y0.x, y0.y);
        af[kk].u[5] = pack_bf16(y0.z, y0.w);
        af[kk].u[6] = pack_bf16(y1.x, y1.y);
        af[kk].u[7] = pack_bf16(y1.z, y1.w);
    }
}

__device__ __forceinline__ float silu_fast(float x) {
    // x * sigmoid(x): v_exp_f32 + v_rcp_f32, no IEEE divide expansion
    return x * __builtin_amdgcn_rcpf(1.0f + __expf(-x));
}

__global__ __launch_bounds__(256) void hadamard_mlp_pairwise_kernel(
    const float* __restrict__ tgt,   // [B,N,D]
    const float* __restrict__ src,   // [B,M,D]
    const float* __restrict__ Win,   // [H,D]
    const float* __restrict__ bin,   // [H]
    const float* __restrict__ Wout,  // [1,H]
    const float* __restrict__ bout,  // [1]
    float* __restrict__ out)         // [B,N,M]
{
    __shared__ float        tval[DIM];
    __shared__ float        bias_lds[HID];
    __shared__ float        wout_lds[HID];
    __shared__ unsigned int amat[HID * ASTRIDE];   // A_i[h][d] as packed bf16 pairs

    const int i   = blockIdx.x;      // target row
    const int b   = blockIdx.y;      // batch
    const int tid = threadIdx.x;

    // ---- stage t_i / b_in / w_out, build A_i = W_in * t_i (bf16) in LDS ----
    const float* trow = tgt + ((size_t)b * NT + i) * DIM;
    if (tid < DIM) {
        tval[tid]     = trow[tid];
        bias_lds[tid] = bin[tid];
        wout_lds[tid] = Wout[tid];
    }
    __syncthreads();

    // 128 rows x 32 float4-quads = 4096 quads / 256 threads = 16 steps
    for (int p = 0; p < 16; ++p) {
        int idx = tid + 256 * p;
        int h  = idx >> 5;
        int c4 = idx & 31;                          // quad column: d = 4*c4 .. 4*c4+3
        float4 w = *(const float4*)(Win + h * DIM + 4 * c4);
        float4 t = *(const float4*)(tval + 4 * c4);
        uint2 pk;
        pk.x = pack_bf16(w.x * t.x, w.y * t.y);
        pk.y = pack_bf16(w.z * t.z, w.w * t.w);
        *(uint2*)(amat + h * ASTRIDE + 2 * c4) = pk;   // 8B aligned
    }
    __syncthreads();

    const int wave = tid >> 5;
    const int lane = tid & 31;
    const int m16  = lane & 15;      // WMMA row/col within 16-lane half
    const int hi   = lane >> 4;      // lane-half select

    const float bo = bout[0];
    const float* sbase = src + (size_t)b * MS * DIM;
    float*       obase = out + ((size_t)b * NT + i) * MS;

    // Each wave owns 16 j-tiles, processed as 8 pairs of adjacent tiles so every
    // B-fragment (2x ds_load_b128) feeds TWO independent WMMA chains.
    for (int pp = 0; pp < 8; ++pp) {
        const int j0 = 32 * wave + 256 * pp;     // rows j0 .. j0+31

        if (pp < 7)
            __builtin_prefetch(sbase + (size_t)(j0 + 256 + m16) * DIM, 0, 3);

        FragBF af0[4], af1[4];
        load_afrags(sbase + (size_t)(j0 + m16) * DIM,      hi, af0);
        load_afrags(sbase + (size_t)(j0 + 16 + m16) * DIM, hi, af1);

        float racc0[8], racc1[8];
        #pragma unroll
        for (int r = 0; r < 8; ++r) { racc0[r] = 0.0f; racc1[r] = 0.0f; }

        for (int ht = 0; ht < 8; ++ht) {
            const float bbv = bias_lds[ht * 16 + m16];
            const float wwv = wout_lds[ht * 16 + m16];

            v8f c0 = {}, c1 = {};
            #pragma unroll
            for (int kk = 0; kk < 4; ++kk) {             // K = 128 in 4 steps of 32
                FragBF bf;                               // B[k][n] = A_i[h0+n][32kk + 2v + 8hi (+16 for v>=4) (+1)]
                const unsigned int* lp =
                    &amat[(ht * 16 + m16) * ASTRIDE + 16 * kk + 4 * hi];  // 16B aligned
                bf.q[0] = *(const uint4*)(lp);
                bf.q[1] = *(const uint4*)(lp + 8);
                c0 = __builtin_amdgcn_wmma_f32_16x16x32_bf16(
                         false, af0[kk].v, false, bf.v, (short)0, c0, false, false);
                c1 = __builtin_amdgcn_wmma_f32_16x16x32_bf16(
                         false, af1[kk].v, false, bf.v, (short)0, c1, false, false);
            }
            // fused epilogue: cX[r] = hidden[j0 (+16) + r + 8*hi][ht*16 + m16]
            #pragma unroll
            for (int r = 0; r < 8; ++r) {
                racc0[r] += wwv * silu_fast(c0[r] + bbv);
                racc1[r] += wwv * silu_fast(c1[r] + bbv);
            }

            // Scheduling fence: WMMAs (0x8) and writes must stay in their h-tile
            // region (bounds live accumulator sets -> VGPRs < 256, no MSB mode);
            // VALU/SALU/TRANS (epilogue) and VMEM/DS *reads* (next tile's operands)
            // may still cross to overlap with the matrix pipe.
            __builtin_amdgcn_sched_barrier(0x2 | 0x4 | 0x20 | 0x100 | 0x400);
        }

        // reduce over the 16 h-columns across each 16-lane half, store 32 outputs
        #pragma unroll
        for (int r = 0; r < 8; ++r) {
            float v0 = racc0[r];
            v0 += __shfl_xor(v0, 1, 32);
            v0 += __shfl_xor(v0, 2, 32);
            v0 += __shfl_xor(v0, 4, 32);
            v0 += __shfl_xor(v0, 8, 32);
            if (m16 == r) obase[j0 + r + 8 * hi] = v0 + bo;

            float v1 = racc1[r];
            v1 += __shfl_xor(v1, 1, 32);
            v1 += __shfl_xor(v1, 2, 32);
            v1 += __shfl_xor(v1, 4, 32);
            v1 += __shfl_xor(v1, 8, 32);
            if (m16 == r) obase[j0 + 16 + r + 8 * hi] = v1 + bo;
        }
    }
}

extern "C" void kernel_launch(void* const* d_in, const int* in_sizes, int n_in,
                              void* d_out, int out_size, void* d_ws, size_t ws_size,
                              hipStream_t stream) {
    (void)in_sizes; (void)n_in; (void)d_ws; (void)ws_size; (void)out_size;
    const float* tgt  = (const float*)d_in[0];   // target_val [B,N,D]
    const float* src  = (const float*)d_in[1];   // source_val [B,M,D]
    const float* Win  = (const float*)d_in[2];   // W_in [H,D]
    const float* bin  = (const float*)d_in[3];   // b_in [H]
    const float* Wout = (const float*)d_in[4];   // W_out [1,H]
    const float* bout = (const float*)d_in[5];   // b_out [1]
    float* out = (float*)d_out;                  // [B,N,M]

    dim3 grid(NT, BATCH);
    dim3 block(256);
    hadamard_mlp_pairwise_kernel<<<grid, block, 0, stream>>>(
        tgt, src, Win, bin, Wout, bout, out);
}
